// trimmed_SchNet_20332375179748
// MI455X (gfx1250) — matrix operand order repaired and checked by reading, not compiled, not run on hardware
//
#include <hip/hip_runtime.h>
#include <cstdint>

// ---------------- problem constants ----------------
constexpr int kNAT = 1024;
constexpr int kDEG = 32;
constexpr int kD   = 128;
constexpr int kR   = 64;
constexpr int kL   = 3;
constexpr int kB   = 16;
constexpr int kE   = kNAT * kDEG;   // 32768 = 2^15
constexpr float kLOG2 = 0.6931471805599453f;

// ---------------- types ----------------
typedef __attribute__((ext_vector_type(16))) _Float16 v16h;
typedef __attribute__((ext_vector_type(2)))  _Float16 v2h;
typedef __attribute__((ext_vector_type(8)))  float    v8f;
typedef __attribute__((ext_vector_type(4)))  unsigned int v4u;
typedef __attribute__((ext_vector_type(8)))  int      v8i;
typedef __attribute__((ext_vector_type(4)))  int      v4i;

// ---------------- WMMA ----------------
static __device__ inline v8f wmma_f16(v16h a, v16h b, v8f c) {
  return __builtin_amdgcn_wmma_f32_16x16x32_f16(
      false, a, false, b, (short)0, c, false, false);
}

// Fragment-swizzled LDS layout: each WMMA fragment occupies 1024 contiguous
// bytes; lane L owns bytes [L*32, L*32+32) == its 16 register halves.
static __device__ inline v16h frag16(const _Float16* base, int frag, int lane) {
  return *(const v16h*)(base + (((frag << 5) + lane) << 4));
}

// A-matrix swizzle (ISA 7.12.2, 16-bit A 16x32)
static __device__ inline int a_swz(int r, int k, int KC) {
  int wv = r >> 4, ml = r & 15;
  int c = k >> 5, kr = k & 31;
  int h = (kr >> 3) & 1;
  int q = kr - (h << 3);
  int j = (q < 8) ? q : q - 8;
  return ((((wv * KC + c) << 5) | (ml | (h << 4))) << 4) | j;
}
// B-matrix swizzle (32x16)
static __device__ inline int b_swz(int k, int n, int NT) {
  int c = k >> 5, kr = k & 31;
  int lane = (n & 15) | (((kr >> 3) & 1) << 4);
  int j = (kr & 7) | ((kr >> 4) << 3);
  int t = n >> 4;
  return ((((c * NT + t) << 5) | lane) << 4) | j;
}

// WMMA block with batched fragment loads: all NT B fragments of a k-chunk
// are loaded into distinct registers (2*NT+2 ds_load_b128 in flight) before
// the NT wmmas issue, so the scheduler emits staggered partial s_wait_dscnt
// instead of a full wait before every wmma.
template <int NT>
static __device__ inline void wmma_block_swz(const _Float16* Aswz, int wv,
                                             int KC, const _Float16* Bswz,
                                             int lane, v8f* acc) {
  v16h af = frag16(Aswz, wv * KC, lane);
  for (int c = 0; c < KC; ++c) {
    v16h bfr[NT];
#pragma unroll
    for (int t = 0; t < NT; ++t) bfr[t] = frag16(Bswz, c * NT + t, lane);
    v16h acur = af;
    const int cn = (c + 1 < KC) ? c + 1 : c;   // prefetch next chunk's A
    af = frag16(Aswz, wv * KC + cn, lane);
#pragma unroll
    for (int t = 0; t < NT; ++t) acc[t] = wmma_f16(acur, bfr[t], acc[t]);
  }
}

// ---------------- TDM bulk copy (global f16, pre-swizzled -> LDS) ----------
#if defined(__HIP_DEVICE_COMPILE__) &&                                   \
    __has_builtin(__builtin_amdgcn_tensor_load_to_lds) &&                \
    __has_builtin(__builtin_amdgcn_s_wait_tensorcnt)
#define USE_TDM 1
#else
#define USE_TDM 0
#endif

#if USE_TDM
static __device__ inline void tdm_load(_Float16* lds, const _Float16* gsrc,
                                       int bytes) {
  unsigned long long ga = (unsigned long long)(uintptr_t)gsrc;
  unsigned int la = (unsigned int)(uintptr_t)lds;  // addr[31:0] = LDS offset
  unsigned int cnt = (unsigned int)(bytes >> 3);   // 8-byte units
  // D# group0: count=1 valid | lds_addr | global_addr | type=2
  v4u g0 = {1u, la, (unsigned int)ga,
            (unsigned int)((ga >> 32) & 0x01FFFFFFu) | 0x80000000u};
  // D# group1: data_size=8B; tensor_dim0=cnt; tensor_dim1=1; tile_dim0=cnt;
  //            tensor_dim0_stride=cnt
  v8i g1 = {(int)(3u << 16),
            (int)((cnt & 0xFFFFu) << 16),
            (int)((cnt >> 16) | (1u << 16)),
            (int)((cnt & 0xFFFFu) << 16),
            0, (int)cnt, 0, 0};
  v4i z4 = {0, 0, 0, 0};
#if __clang_major__ >= 23
  v8i z8 = {0, 0, 0, 0, 0, 0, 0, 0};
  __builtin_amdgcn_tensor_load_to_lds(g0, g1, z4, z4, z8, 0);
#else
  __builtin_amdgcn_tensor_load_to_lds(g0, g1, z4, z4, 0);
#endif
}
#endif

// Copy `halves` f16 (contiguous) from global to LDS. Caller barriers after.
static __device__ inline void copy_w_lds(_Float16* dst, const _Float16* src,
                                         int halves, int tid) {
#if USE_TDM
  if (tid < 32) {                       // one wave issues the DMA
    tdm_load(dst, src, halves * 2);
    __builtin_amdgcn_s_wait_tensorcnt(0);
  }
#else
  for (int i = tid; i < (halves >> 4); i += 256)
    ((v16h*)dst)[i] = ((const v16h*)src)[i];
#endif
}

// ---------------- math helpers ----------------
static __device__ inline float dssp(float x) {  // softplus(x) - log2
  float sp = (x > 20.f) ? x : log1pf(expf(x));
  return sp - kLOG2;
}
static __device__ inline float dsig(float x) { return 1.f / (1.f + expf(-x)); }

// =====================================================================
// Weight pre-swizzle: f32 global -> f16 fragment-swizzled global view.
// =====================================================================
__global__ void wswz_k(const float* __restrict__ W, _Float16* __restrict__ out,
                       int K, int N, int transB) {
  const int NT = N >> 4;
  const int kshift = (K == 128) ? 7 : 6;
  const int nshift = (N == 128) ? 7 : 6;
  for (int idx = blockIdx.x * blockDim.x + threadIdx.x; idx < K * N;
       idx += gridDim.x * blockDim.x) {
    int k, n;
    float w;
    if (transB) { n = idx >> kshift; k = idx & (K - 1); w = W[n * K + k]; }
    else        { k = idx >> nshift; n = idx & (N - 1); w = W[k * N + n]; }
    out[b_swz(k, n, NT)] = (_Float16)w;
  }
}

// =====================================================================
// GEMM: C[M x N] = epi( actIn(A)[M x K] @ Wswz[K x N] )
// =====================================================================
template <int NT>
__global__ __launch_bounds__(256) void gemm_k(
    const float* __restrict__ A, const _Float16* __restrict__ Wswz,
    const float* __restrict__ bias, const float* __restrict__ aux,
    const float* __restrict__ resid, float* __restrict__ C,
    int N, int K, int actInSSP) {
  extern __shared__ char smem[];
  const int KC = K >> 5;
  _Float16* Aswz = (_Float16*)smem;          // 128*K halves
  _Float16* Bswz = Aswz + 128 * K;           // K*N halves

  const int tid = threadIdx.x;
  const int blockRow = blockIdx.x * 128;
  const int kshift = (K == 128) ? 7 : 6;

  copy_w_lds(Bswz, Wswz, K * N, tid);
  for (int idx = tid; idx < 64 * K; idx += 256) {
    int r = idx >> (kshift - 1);
    int c = (idx & ((K >> 1) - 1)) << 1;
    const float* ap = A + (blockRow + r) * K + c;
    float v0 = ap[0], v1 = ap[1];
    if (actInSSP) { v0 = dssp(v0); v1 = dssp(v1); }
    v2h p; p[0] = (_Float16)v0; p[1] = (_Float16)v1;
    *(v2h*)(Aswz + a_swz(r, c, KC)) = p;
  }
  __syncthreads();

  const int lane = tid & 31, wave = tid >> 5;
  v8f acc[NT] = {};
  wmma_block_swz<NT>(Aswz, wave, KC, Bswz, lane, acc);

  const int laneLo = lane & 15, laneHi = lane >> 4;
#pragma unroll
  for (int t = 0; t < NT; ++t)
#pragma unroll
    for (int v = 0; v < 8; ++v) {
      int m = blockRow + wave * 16 + v + 8 * laneHi;
      int col = t * 16 + laneLo;
      float val = acc[t][v];
      if (bias)  val += bias[col];
      if (aux)   val *= dsig(aux[m * N + col]);
      if (resid) val += resid[m * N + col];
      C[m * N + col] = val;
    }
}

// =====================================================================
// Edge-MLP forward fused with segment-sum (128 edges == 4 atoms per block).
// =====================================================================
__global__ __launch_bounds__(256) void edge_fwd_k(
    const float* __restrict__ dist, const int* __restrict__ idx_j,
    const float* __restrict__ x1, const _Float16* __restrict__ sWf1,
    const float* __restrict__ bf1, const _Float16* __restrict__ sWf2,
    const float* __restrict__ bf2, const float* __restrict__ centers,
    const float* __restrict__ gamma, float* __restrict__ y) {
  extern __shared__ char smem[];
  float* dsh  = (float*)smem;                       // 128
  float* yacc = dsh + 128;                          // 512
  int*   xoff = (int*)(yacc + 512);                 // 128  x1 row base
  float* csh  = (float*)(xoff + 128);               // 64
  float* gsh  = csh + 64;                           // 64
  _Float16* rbfs = (_Float16*)(gsh + 64);           // 128x64  A-swz (KC=2)
  _Float16* Wsh  = rbfs + 128 * kR;                 // up to 128x128 B-swz
  _Float16* t1s  = Wsh + kD * kD;                   // 128x128 A-swz (KC=4)

  const int tid = threadIdx.x, lane = tid & 31, wave = tid >> 5;
  const int ge0 = blockIdx.x * 128;

  for (int r = tid; r < 128; r += 256) {
    int e = ge0 + r;
    dsh[r] = dist[e];
    xoff[r] = (((e >> 15) << 10) + idx_j[e & (kE - 1)]) << 7;
  }
  if (tid < 64) { csh[tid] = centers[tid]; gsh[tid] = gamma[tid]; }
  for (int i = tid; i < 512; i += 256) yacc[i] = 0.f;
  copy_w_lds(Wsh, sWf1, kR * kD, tid);
  __syncthreads();
  for (int idx = tid; idx < 128 * 32; idx += 256) {   // rbf pairs
    int r = idx >> 5, q = (idx & 31) << 1;
    float dl = dsh[r];
    float t0 = dl - csh[q],     g0 = gsh[q];
    float t1 = dl - csh[q + 1], g1 = gsh[q + 1];
    v2h p; p[0] = (_Float16)expf(-g0 * t0 * t0);
    p[1] = (_Float16)expf(-g1 * t1 * t1);
    *(v2h*)(rbfs + a_swz(r, q, 2)) = p;
  }
  __syncthreads();

  const int laneLo = lane & 15, laneHi = lane >> 4;
  v8f acc1[8] = {};
  wmma_block_swz<8>(rbfs, wave, 2, Wsh, lane, acc1);  // u1 = rbf @ Wf1
  if (tid < 32) __builtin_prefetch(sWf2, 0, 1);
  __syncthreads();
#pragma unroll
  for (int t = 0; t < 8; ++t)
#pragma unroll
    for (int v = 0; v < 8; ++v) {
      int m = wave * 16 + v + 8 * laneHi, col = t * 16 + laneLo;
      t1s[a_swz(m, col, 4)] = (_Float16)dssp(acc1[t][v] + bf1[col]);
    }
  copy_w_lds(Wsh, sWf2, kD * kD, tid);
  __syncthreads();

  v8f acc2[8] = {};
  wmma_block_swz<8>(t1s, wave, 4, Wsh, lane, acc2);   // u2 = t1 @ Wf2
#pragma unroll
  for (int t = 0; t < 8; ++t)
#pragma unroll
    for (int v = 0; v < 8; ++v) {
      int m = wave * 16 + v + 8 * laneHi, col = t * 16 + laneLo;
      float f = dssp(acc2[t][v] + bf2[col]);
      float xv = x1[xoff[m] + col];
      atomicAdd(&yacc[((m >> 5) << 7) + col], f * xv);
    }
  __syncthreads();
  const int af0 = ge0 >> 5;  // flat-edge>>5 == flat-atom
  for (int i = tid; i < 512; i += 256)
    y[((af0 + (i >> 7)) << 7) + (i & 127)] = yacc[i];
}

// =====================================================================
// Edge-MLP backward (recompute; checkpoint is d only).
// =====================================================================
__global__ __launch_bounds__(256) void edge_bwd_k(
    const float* __restrict__ dist, const int* __restrict__ idx_j,
    const float* __restrict__ x1, const float* __restrict__ gy,
    const _Float16* __restrict__ sWf1, const float* __restrict__ bf1,
    const _Float16* __restrict__ sWf2, const float* __restrict__ bf2,
    const _Float16* __restrict__ sWf2T, const _Float16* __restrict__ sWf1T,
    const float* __restrict__ centers, const float* __restrict__ gamma,
    float* __restrict__ gx, float* __restrict__ gd) {
  extern __shared__ char smem[];
  float* dsh   = (float*)smem;                      // 128
  float* gdacc = dsh + 128;                         // 128
  int*   xoff  = (int*)(gdacc + 128);               // 128  x1/gx row base
  int*   goff  = xoff + 128;                        // 128  gy row base
  float* csh   = (float*)(goff + 128);              // 64
  float* gsh   = csh + 64;                          // 64
  _Float16* rbfs = (_Float16*)(gsh + 64);           // 128x64  A-swz
  _Float16* Wsh  = rbfs + 128 * kR;                 // B-swz region
  _Float16* R2   = Wsh + kD * kD;                   // t1 -> du2   (A-swz KC=4)
  _Float16* R3   = R2 + kD * kD;                    // sig(u1) -> dt1

  const int tid = threadIdx.x, lane = tid & 31, wave = tid >> 5;
  const int ge0 = blockIdx.x * 128;
  const int laneLo = lane & 15, laneHi = lane >> 4;

  for (int r = tid; r < 128; r += 256) {
    int e = ge0 + r, b = e >> 15, el = e & (kE - 1);
    dsh[r] = dist[e];
    xoff[r] = (((b << 10) + idx_j[el]) << 7);
    goff[r] = (((b << 10) + (el >> 5)) << 7);
    gdacc[r] = 0.f;
  }
  if (tid < 64) { csh[tid] = centers[tid]; gsh[tid] = gamma[tid]; }
  copy_w_lds(Wsh, sWf1, kR * kD, tid);
  __syncthreads();
  for (int idx = tid; idx < 128 * 32; idx += 256) {
    int r = idx >> 5, q = (idx & 31) << 1;
    float dl = dsh[r];
    float t0 = dl - csh[q],     g0 = gsh[q];
    float t1 = dl - csh[q + 1], g1 = gsh[q + 1];
    v2h p; p[0] = (_Float16)expf(-g0 * t0 * t0);
    p[1] = (_Float16)expf(-g1 * t1 * t1);
    *(v2h*)(rbfs + a_swz(r, q, 2)) = p;
  }
  __syncthreads();

  // ---- A: u1 ----
  v8f acc1[8] = {};
  wmma_block_swz<8>(rbfs, wave, 2, Wsh, lane, acc1);
  if (tid < 32) __builtin_prefetch(sWf2, 0, 1);
  __syncthreads();
#pragma unroll
  for (int t = 0; t < 8; ++t)
#pragma unroll
    for (int v = 0; v < 8; ++v) {
      int m = wave * 16 + v + 8 * laneHi, col = t * 16 + laneLo;
      int ix = a_swz(m, col, 4);
      float u1 = acc1[t][v] + bf1[col];
      R2[ix] = (_Float16)dssp(u1);
      R3[ix] = (_Float16)dsig(u1);
    }
  copy_w_lds(Wsh, sWf2, kD * kD, tid);
  __syncthreads();

  // ---- B: u2, scatter gx, du2 ----
  v8f acc2[8] = {};
  wmma_block_swz<8>(R2, wave, 4, Wsh, lane, acc2);
  if (tid < 32) __builtin_prefetch(sWf2T, 0, 1);
  __syncthreads();
#pragma unroll
  for (int t = 0; t < 8; ++t)
#pragma unroll
    for (int v = 0; v < 8; ++v) {
      int m = wave * 16 + v + 8 * laneHi, col = t * 16 + laneLo;
      float u2 = acc2[t][v] + bf2[col];
      float f = dssp(u2);
      float gm = gy[goff[m] + col];
      float xv = x1[xoff[m] + col];
      atomicAdd(&gx[xoff[m] + col], gm * f);
      R2[a_swz(m, col, 4)] = (_Float16)(gm * xv * dsig(u2));  // du2
    }
  copy_w_lds(Wsh, sWf2T, kD * kD, tid);
  __syncthreads();

  // ---- C: dt1 = (du2 @ Wf2^T) * sig(u1), in place over R3 ----
  v8f acc3[8] = {};
  wmma_block_swz<8>(R2, wave, 4, Wsh, lane, acc3);
  if (tid < 32) __builtin_prefetch(sWf1T, 0, 1);
  __syncthreads();
#pragma unroll
  for (int t = 0; t < 8; ++t)
#pragma unroll
    for (int v = 0; v < 8; ++v) {
      int m = wave * 16 + v + 8 * laneHi, col = t * 16 + laneLo;
      int ix = a_swz(m, col, 4);
      float s1 = (float)R3[ix];
      R3[ix] = (_Float16)(acc3[t][v] * s1);
    }
  copy_w_lds(Wsh, sWf1T, kD * kR, tid);
  __syncthreads();

  // ---- D: drbf = dt1 @ Wf1^T ; reduce -> gd ----
  v8f acc4[4] = {};
  wmma_block_swz<4>(R3, wave, 4, Wsh, lane, acc4);
#pragma unroll
  for (int t = 0; t < 4; ++t)
#pragma unroll
    for (int v = 0; v < 8; ++v) {
      int m = wave * 16 + v + 8 * laneHi, r = t * 16 + laneLo;
      float tt = dsh[m] - csh[r], g = gsh[r];
      float w = expf(-g * tt * tt) * (-2.f * g * tt);
      atomicAdd(&gdacc[m], acc4[t][v] * w);
    }
  __syncthreads();
  for (int r = tid; r < 128; r += 256) gd[ge0 + r] += gdacc[r];
}

// ---------------- small elementwise kernels ----------------
__global__ void dist_k(const float* __restrict__ pos,
                       const int* __restrict__ idx_j, float* __restrict__ dist) {
  for (int idx = blockIdx.x * blockDim.x + threadIdx.x; idx < kB * kE;
       idx += gridDim.x * blockDim.x) {
    int b = idx >> 15, el = idx & (kE - 1);
    int i = el >> 5, j = idx_j[el];
    const float* pb = pos + b * kNAT * 3;
    float dx = pb[i * 3 + 0] - pb[j * 3 + 0];
    float dy = pb[i * 3 + 1] - pb[j * 3 + 1];
    float dz = pb[i * 3 + 2] - pb[j * 3 + 2];
    dist[idx] = sqrtf(dx * dx + dy * dy + dz * dz + 1e-8f);
  }
}
__global__ void embed_k(const int* __restrict__ types,
                        const float* __restrict__ emb, float* __restrict__ feat) {
  for (int idx = blockIdx.x * blockDim.x + threadIdx.x; idx < kB * kNAT * kD;
       idx += gridDim.x * blockDim.x) {
    int a = idx >> 7, c = idx & 127;
    feat[idx] = emb[types[a] * kD + c];
  }
}
__global__ void zero_k(float* __restrict__ p, int n) {
  for (int i = blockIdx.x * blockDim.x + threadIdx.x; i < n;
       i += gridDim.x * blockDim.x)
    p[i] = 0.f;
}
__global__ void rt_k(const float* __restrict__ ad, const float* __restrict__ We,
                     float* __restrict__ rt) {
  for (int idx = blockIdx.x * blockDim.x + threadIdx.x; idx < kB * kNAT * 64;
       idx += gridDim.x * blockDim.x)
    rt[idx] = We[idx & 63] * dsig(ad[idx]);
}
__global__ void force_k(const float* __restrict__ pos,
                        const int* __restrict__ idx_j,
                        const float* __restrict__ dist,
                        const float* __restrict__ gd, float* __restrict__ out) {
  __shared__ float acc[3];
  int b = blockIdx.x, tid = threadIdx.x;
  if (tid < 3) acc[tid] = 0.f;
  __syncthreads();
  const float* pb = pos + b * kNAT * 3;
  float fx = 0.f, fy = 0.f, fz = 0.f;
  for (int el = tid; el < kE; el += blockDim.x) {
    int i = el >> 5, j = idx_j[el];
    if (i == 0 || j == 0) {
      float g = gd[b * kE + el] / dist[b * kE + el];
      float dx = pb[i * 3 + 0] - pb[j * 3 + 0];
      float dy = pb[i * 3 + 1] - pb[j * 3 + 1];
      float dz = pb[i * 3 + 2] - pb[j * 3 + 2];
      if (i == 0) { fx += g * dx; fy += g * dy; fz += g * dz; }
      if (j == 0) { fx -= g * dx; fy -= g * dy; fz -= g * dz; }
    }
  }
  atomicAdd(&acc[0], fx); atomicAdd(&acc[1], fy); atomicAdd(&acc[2], fz);
  __syncthreads();
  if (tid < 3) out[b * 3 + tid] = -acc[tid];
}

// =====================================================================
// Host orchestration
// =====================================================================
extern "C" void kernel_launch(void* const* d_in, const int* in_sizes, int n_in,
                              void* d_out, int out_size, void* d_ws,
                              size_t ws_size, hipStream_t stream) {
  (void)in_sizes; (void)n_in; (void)out_size; (void)ws_size;
  const float* pos   = (const float*)d_in[0];
  const int*   types = (const int*)d_in[1];
  const int*   idx_j = (const int*)d_in[3];
  const float* emb   = (const float*)d_in[5];
  const float* W1    = (const float*)d_in[6];
  const float* b1    = (const float*)d_in[7];
  const float* Wf1   = (const float*)d_in[8];
  const float* bf1   = (const float*)d_in[9];
  const float* Wf2   = (const float*)d_in[10];
  const float* bf2   = (const float*)d_in[11];
  const float* W2    = (const float*)d_in[12];
  const float* b2    = (const float*)d_in[13];
  const float* W3    = (const float*)d_in[14];
  const float* b3    = (const float*)d_in[15];
  const float* cen   = (const float*)d_in[16];
  const float* gam   = (const float*)d_in[17];
  const float* Wd    = (const float*)d_in[18];
  const float* bd    = (const float*)d_in[19];
  const float* We    = (const float*)d_in[20];
  float* out = (float*)d_out;

  const int A = kB * kNAT * kD;  // 2,097,152 floats
  float* w = (float*)d_ws;
  float* dist = w;               w += kB * kE;
  float* feat[4];
  for (int l = 0; l < 4; ++l) { feat[l] = w; w += A; }
  float* a2c[3];
  for (int l = 0; l < 3; ++l) { a2c[l] = w; w += A; }
  float* adb = w;  w += kB * kNAT * 64;
  float* x1b = w;  w += A;
  float* yb  = w;  w += A;
  float* tmp = w;  w += A;
  float* gyb = w;  w += A;
  float* gxb = w;  w += A;
  float* gA  = w;  w += A;
  float* gB  = w;  w += A;
  float* rtb = w;  w += kB * kNAT * 64;
  float* gdb = w;  w += kB * kE;

  // pre-swizzled f16 weight views
  _Float16* hw = (_Float16*)w;
  auto halloc = [&](int halves) { _Float16* p = hw; hw += halves; return p; };
  _Float16 *sW1[3], *sW2[3], *sW3[3], *sW1T[3], *sW2T[3], *sW3T[3];
  _Float16 *sWf1[3], *sWf2[3], *sWf2T[3], *sWf1T[3];
  for (int l = 0; l < 3; ++l) {
    sW1[l]   = halloc(kD * kD);
    sW2[l]   = halloc(kD * kD);
    sW3[l]   = halloc(kD * kD);
    sW1T[l]  = halloc(kD * kD);
    sW2T[l]  = halloc(kD * kD);
    sW3T[l]  = halloc(kD * kD);
    sWf1[l]  = halloc(kR * kD);
    sWf2[l]  = halloc(kD * kD);
    sWf2T[l] = halloc(kD * kD);
    sWf1T[l] = halloc(kD * kR);
  }
  _Float16* sWd  = halloc(kD * kR);
  _Float16* sWdT = halloc(kR * kD);

  const int MROW = kB * kNAT;  // 16384
  const dim3 gGemm(MROW / 128), gEdge(kB * kE / 128), blk(256);
  auto gemm_sh = [](int K, int N) { return (size_t)(128 * K + K * N) * 2; };
  const size_t shG88 = gemm_sh(128, 128), shG84 = gemm_sh(128, 64),
               shG48 = gemm_sh(64, 128);
  const size_t shEF = 896 * 4 + (size_t)(128 * kR + kD * kD + kD * kD) * 2;
  const size_t shEB =
      640 * 4 + (size_t)(128 * kR + kD * kD + kD * kD + kD * kD) * 2;

  // ---------------- weight prep ----------------
  for (int l = 0; l < 3; ++l) {
    wswz_k<<<64, blk, 0, stream>>>(W1 + l * kD * kD, sW1[l], 128, 128, 0);
    wswz_k<<<64, blk, 0, stream>>>(W2 + l * kD * kD, sW2[l], 128, 128, 0);
    wswz_k<<<64, blk, 0, stream>>>(W3 + l * kD * kD, sW3[l], 128, 128, 0);
    wswz_k<<<64, blk, 0, stream>>>(W1 + l * kD * kD, sW1T[l], 128, 128, 1);
    wswz_k<<<64, blk, 0, stream>>>(W2 + l * kD * kD, sW2T[l], 128, 128, 1);
    wswz_k<<<64, blk, 0, stream>>>(W3 + l * kD * kD, sW3T[l], 128, 128, 1);
    wswz_k<<<32, blk, 0, stream>>>(Wf1 + l * kR * kD, sWf1[l], 64, 128, 0);
    wswz_k<<<64, blk, 0, stream>>>(Wf2 + l * kD * kD, sWf2[l], 128, 128, 0);
    wswz_k<<<64, blk, 0, stream>>>(Wf2 + l * kD * kD, sWf2T[l], 128, 128, 1);
    wswz_k<<<32, blk, 0, stream>>>(Wf1 + l * kR * kD, sWf1T[l], 128, 64, 1);
  }
  wswz_k<<<32, blk, 0, stream>>>(Wd, sWd, 128, 64, 0);
  wswz_k<<<32, blk, 0, stream>>>(Wd, sWdT, 64, 128, 1);

  // ---------------- forward ----------------
  dist_k<<<2048, blk, 0, stream>>>(pos, idx_j, dist);
  embed_k<<<4096, blk, 0, stream>>>(types, emb, feat[0]);
  for (int l = 0; l < kL; ++l) {
    gemm_k<8><<<gGemm, blk, shG88, stream>>>(feat[l], sW1[l], b1 + l * kD,
        nullptr, nullptr, x1b, kD, kD, 0);
    edge_fwd_k<<<gEdge, blk, shEF, stream>>>(dist, idx_j, x1b, sWf1[l],
        bf1 + l * kD, sWf2[l], bf2 + l * kD, cen + l * kR, gam + l * kR, yb);
    gemm_k<8><<<gGemm, blk, shG88, stream>>>(yb, sW2[l], b2 + l * kD,
        nullptr, nullptr, a2c[l], kD, kD, 0);
    gemm_k<8><<<gGemm, blk, shG88, stream>>>(a2c[l], sW3[l], b3 + l * kD,
        nullptr, feat[l], feat[l + 1], kD, kD, 1);
  }
  gemm_k<4><<<gGemm, blk, shG84, stream>>>(feat[3], sWd, bd, nullptr, nullptr,
                                           adb, 64, kD, 0);

  // ---------------- backward ----------------
  rt_k<<<4096, blk, 0, stream>>>(adb, We, rtb);
  gemm_k<8><<<gGemm, blk, shG48, stream>>>(rtb, sWdT, nullptr, nullptr, nullptr,
                                           gA, kD, 64, 0);
  zero_k<<<512, blk, 0, stream>>>(gdb, kB * kE);

  float* gcur = gA;
  float* gnext = gB;
  for (int l = kL - 1; l >= 0; --l) {
    gemm_k<8><<<gGemm, blk, shG88, stream>>>(gcur, sW3T[l], nullptr, a2c[l],
        nullptr, tmp, kD, kD, 0);                   // da2
    gemm_k<8><<<gGemm, blk, shG88, stream>>>(tmp, sW2T[l], nullptr, nullptr,
        nullptr, gyb, kD, kD, 0);                   // gy
    gemm_k<8><<<gGemm, blk, shG88, stream>>>(feat[l], sW1[l], b1 + l * kD,
        nullptr, nullptr, x1b, kD, kD, 0);          // x1 recompute
    zero_k<<<4096, blk, 0, stream>>>(gxb, A);
    edge_bwd_k<<<gEdge, blk, shEB, stream>>>(dist, idx_j, x1b, gyb, sWf1[l],
        bf1 + l * kD, sWf2[l], bf2 + l * kD, sWf2T[l], sWf1T[l],
        cen + l * kR, gam + l * kR, gxb, gdb);
    gemm_k<8><<<gGemm, blk, shG88, stream>>>(gxb, sW1T[l], nullptr, nullptr,
        gcur, gnext, kD, kD, 0);                    // gfeat_l
    float* t = gcur; gcur = gnext; gnext = t;
  }
  force_k<<<kB, blk, 0, stream>>>(pos, idx_j, dist, gdb, out);
}
